// MultiHeadLocalAttention_55800215109727
// MI455X (gfx1250) — matrix operand order repaired and checked by reading, not compiled, not run on hardware
//
#include <hip/hip_runtime.h>
#include <hip/hip_bf16.h>

// Problem constants (match reference)
#define BQ   2
#define SQ   8192
#define FQ   512
#define HQ   8
#define DQ   64
#define BLKQ 128
#define NBQ  (SQ / BLKQ)      // 64
#define MROWS (BQ * SQ)       // 16384

typedef __attribute__((ext_vector_type(16))) _Float16 v16h;
typedef __attribute__((ext_vector_type(8)))  _Float16 v8h;
typedef __attribute__((ext_vector_type(8)))  float    v8f;
typedef __attribute__((ext_vector_type(4)))  unsigned su4;
typedef __attribute__((ext_vector_type(8)))  unsigned su8;

// ---- ASYNCcnt wait (builtin if available, else raw asm) ----
#if defined(__has_builtin)
#  if __has_builtin(__builtin_amdgcn_s_wait_asynccnt)
#    define WAIT_ASYNC0() __builtin_amdgcn_s_wait_asynccnt(0)
#  endif
#endif
#ifndef WAIT_ASYNC0
#  define WAIT_ASYNC0() asm volatile("s_wait_asynccnt 0x0" ::: "memory")
#endif

static __device__ inline v8f vzero8() {
  v8f v;
#pragma unroll
  for (int i = 0; i < 8; ++i) v[i] = 0.0f;
  return v;
}

// Async 16-byte global -> LDS copy (per lane), tracked with ASYNCcnt.
static __device__ inline void async_cp16(_Float16* lds_dst, const _Float16* gsrc) {
  asm volatile("global_load_async_to_lds_b128 %0, %1, off"
               :: "v"((unsigned)(uintptr_t)lds_dst),
                  "v"((unsigned long long)(uintptr_t)gsrc)
               : "memory");
}

// TDM: DMA a [rows x 64-half] K tile (row stride 64 halfs in memory) into LDS,
// padding each 32-DWORD row with 4 DWORDs -> LDS row stride 72 halfs.
static __device__ inline void tdm_load_k(const _Float16* gsrc, _Float16* lds_dst) {
  const unsigned long long gK = (unsigned long long)(uintptr_t)gsrc;
  su4 g0;
  g0[0] = 1u;                                   // count=1, user mode
  g0[1] = (unsigned)(uintptr_t)lds_dst;         // lds_addr (bytes)
  g0[2] = (unsigned)gK;                         // global_addr[31:0]
  g0[3] = ((unsigned)(gK >> 32) & 0x01ffffffu)  // global_addr[56:32]
          | (2u << 30);                         // type = 2 ("image")
  su8 g1;
  g1[0] = (1u << 16)      // data_size = 1 (2 bytes)
        | (1u << 20)      // pad_enable
        | (4u << 22)      // pad_interval = 32 DWORDs (one 64-half row)
        | (3u << 25);     // pad_amount   = 4 DWORDs (8 halfs) -> stride 72
  g1[1] = (unsigned)DQ << 16;        // tensor_dim0 = 64   (bits 63:48)
  g1[2] = (unsigned)BLKQ << 16;      // tensor_dim1 = 128  (bits 95:80)
  g1[3] = (unsigned)DQ << 16;        // tile_dim0  = 64    (bits 127:112)
  g1[4] = (unsigned)BLKQ;            // tile_dim1  = 128   (bits 143:128)
  g1[5] = (unsigned)DQ;              // tensor_dim0_stride = 64
  g1[6] = 0u;
  g1[7] = 0u;
  asm volatile("tensor_load_to_lds %0, %1" :: "s"(g0), "s"(g1) : "memory");
}

// A-matrix 16x32 f16 fragment (CDNA5 layout): lane holds row (lane&15);
// halves 0..7 = k (lane>=16?8:0)+0..7 ; halves 8..15 = k 16+(lane>=16?8:0)+0..7
static __device__ inline v16h ld_afrag(const _Float16* base, int stride, int lane) {
  const _Float16* p = base + (lane & 15) * stride + ((lane >> 4) << 3);
  v16h a;
  v8h* h = (v8h*)&a;
  h[0] = *(const v8h*)(p);
  h[1] = *(const v8h*)(p + 16);
  return a;
}

// B-matrix 32x16 f16 fragment: lane holds column (lane&15);
// halves 0..7 = k (lane>=16?8:0)+0..7 ; halves 8..15 = k 16+(lane>=16?8:0)+0..7 ("Bt" layout).
static __device__ inline v16h ld_bfrag(const _Float16* base, int stride, int lane) {
  const _Float16* p = base + (lane & 15) * stride + ((lane >> 4) << 3);
  v16h b;
  v8h* h = (v8h*)&b;
  h[0] = *(const v8h*)(p);
  h[1] = *(const v8h*)(p + 16);
  return b;
}

static __device__ inline void load16f(float* r, const float* p) {
#pragma unroll
  for (int i = 0; i < 16; i += 4) {
    float4 f = *(const float4*)(p + i);
    r[i + 0] = f.x; r[i + 1] = f.y; r[i + 2] = f.z; r[i + 3] = f.w;
  }
}

// ---------------------------------------------------------------------------
// Stage A: fused QKV projection.  C = X[16384,512] x W[512,512] + bias.
// z=0: Q -> f16 [B,H,S,D]; z=1: K -> f16 [B,H,S,D]; z=2: V -> f16 [B,H,D,S]
// Software pipeline: global loads of chunk kc+1 overlap WMMAs on chunk kc;
// LDS tiles double-buffered, one barrier per iteration.
// ---------------------------------------------------------------------------
__global__ __launch_bounds__(256)
void qkv_proj_kernel(const float* __restrict__ Xq, const float* __restrict__ Xkv,
                     const float* __restrict__ Wq, const float* __restrict__ bq,
                     const float* __restrict__ Wk, const float* __restrict__ bk,
                     const float* __restrict__ Wv, const float* __restrict__ bv,
                     _Float16* __restrict__ Qh, _Float16* __restrict__ Kh,
                     _Float16* __restrict__ Vth) {
  constexpr int LDK = 40;                       // padded row stride (halfs)
  constexpr int TSZ = 128 * LDK;                // one tile buffer
  __shared__ _Float16 As[2 * TSZ];              // [buf][row][k]  (A layout)
  __shared__ _Float16 Bs[2 * TSZ];              // [buf][n][k]    (transposed B)

  const int z = blockIdx.z;
  const float* X    = (z == 0) ? Xq : Xkv;
  const float* W    = (z == 0) ? Wq : (z == 1 ? Wk : Wv);
  const float* bias = (z == 0) ? bq : (z == 1 ? bk : bv);
  _Float16* Out     = (z == 0) ? Qh : (z == 1 ? Kh : Vth);
  const bool vtrans = (z == 2);

  const int m0 = blockIdx.x * 128;
  const int n0 = blockIdx.y * 128;
  const int tid = threadIdx.x;
  const int lane = tid & 31;
  const int w  = tid >> 5;
  const int wm = w & 3;                         // 4 strips of 32 rows
  const int wn = w >> 2;                        // 2 strips of 64 cols

  // staging coordinates
  const int arow = tid >> 1, ac0 = (tid & 1) * 16;
  const int bk_ = tid >> 3,  bn0 = (tid & 7) * 16;
  const float* aptr = X + (size_t)(m0 + arow) * FQ + ac0;
  const float* bptr = W + (size_t)bk_ * FQ + n0 + bn0;

  v8f acc[2][4];
#pragma unroll
  for (int i = 0; i < 2; ++i)
#pragma unroll
    for (int j = 0; j < 4; ++j) acc[i][j] = vzero8();

  float areg[16], breg[16];

  // prologue: fetch + stage chunk 0 into buffer 0
  load16f(areg, aptr);
  load16f(breg, bptr);
  {
    v8h h0, h1;
#pragma unroll
    for (int i = 0; i < 8; ++i) { h0[i] = (_Float16)areg[i]; h1[i] = (_Float16)areg[8 + i]; }
    _Float16* dst = As + arow * LDK + ac0;
    *(v8h*)(dst) = h0;
    *(v8h*)(dst + 8) = h1;
#pragma unroll
    for (int i = 0; i < 16; ++i) Bs[(bn0 + i) * LDK + bk_] = (_Float16)breg[i];
  }

  int cur = 0;
  for (int kc = 0; kc < FQ; kc += 32) {
    __syncthreads();                            // buffer `cur` visible to all
    const bool have_next = (kc + 32 < FQ);
    if (have_next) {                            // issue next global loads early
      __builtin_prefetch(aptr + kc + 64, 0, 3);
      load16f(areg, aptr + kc + 32);
      load16f(breg, bptr + (size_t)(kc + 32) * FQ);
    }

    // ---- WMMAs on buffer `cur` ----
    const _Float16* Ab = As + cur * TSZ;
    const _Float16* Bb = Bs + cur * TSZ;
    v16h bfr[4];
#pragma unroll
    for (int nt = 0; nt < 4; ++nt)
      bfr[nt] = ld_bfrag(Bb + (wn * 64 + nt * 16) * LDK, LDK, lane);
#pragma unroll
    for (int mt = 0; mt < 2; ++mt) {
      v16h afr = ld_afrag(Ab + (wm * 32 + mt * 16) * LDK, LDK, lane);
#pragma unroll
      for (int nt = 0; nt < 4; ++nt)
        acc[mt][nt] = __builtin_amdgcn_wmma_f32_16x16x32_f16(
            false, afr, false, bfr[nt], (short)0, acc[mt][nt], false, false);
    }

    // ---- stage next chunk into buffer cur^1 ----
    if (have_next) {
      _Float16* Ad = As + (cur ^ 1) * TSZ + arow * LDK + ac0;
      _Float16* Bd = Bs + (cur ^ 1) * TSZ;
      v8h h0, h1;
#pragma unroll
      for (int i = 0; i < 8; ++i) { h0[i] = (_Float16)areg[i]; h1[i] = (_Float16)areg[8 + i]; }
      *(v8h*)(Ad) = h0;
      *(v8h*)(Ad + 8) = h1;
#pragma unroll
      for (int i = 0; i < 16; ++i) Bd[(bn0 + i) * LDK + bk_] = (_Float16)breg[i];
    }
    cur ^= 1;
  }

  // ---- epilogue: bias add, f32->f16, scatter to head layout ----
#pragma unroll
  for (int nt = 0; nt < 4; ++nt) {
    const int n = n0 + wn * 64 + nt * 16 + (lane & 15);
    const float bv_ = bias[n];
    const int hh = n >> 6;
    const int dd = n & 63;
#pragma unroll
    for (int mt = 0; mt < 2; ++mt) {
#pragma unroll
      for (int r = 0; r < 8; ++r) {
        const int m = m0 + wm * 32 + mt * 16 + ((lane >> 4) << 3) + r;
        const int bb = m >> 13;                 // / SQ
        const int ss = m & (SQ - 1);
        const size_t addr = vtrans
            ? ((size_t)((bb * HQ + hh) * DQ + dd) * SQ + ss)
            : ((size_t)((bb * HQ + hh) * SQ + ss) * DQ + dd);
        Out[addr] = (_Float16)(acc[mt][nt][r] + bv_);
      }
    }
  }
}

// ---------------------------------------------------------------------------
// Stage B: block-local attention with online (flash) softmax.
// K tile: Tensor Data Mover (descriptor, LDS pad).  V^T tile: async b128.
// Double-buffered so block jb+1's DMA overlaps block jb's WMMAs/softmax.
// ---------------------------------------------------------------------------
__global__ __launch_bounds__(256)
void local_attn_kernel(const _Float16* __restrict__ Qh, const _Float16* __restrict__ Kh,
                       const _Float16* __restrict__ Vth, _Float16* __restrict__ Oh) {
  constexpr int KLD = 72;    // Ks row stride (halfs) = 64 + TDM pad
  constexpr int VLD = 136;   // Vs row stride
  constexpr int PLD = 136;   // P row stride
  constexpr int KSZ = BLKQ * KLD;
  constexpr int VSZ = DQ * VLD;
  __shared__ _Float16 Ks[2 * KSZ];          // K block, [buf][key][d]
  __shared__ _Float16 Vs[2 * VSZ];          // V^T block, [buf][d][key]
  __shared__ _Float16 Ps[8 * 16 * PLD];     // per-wave P tile, [row][key]

  const int idx = blockIdx.x;
  const int n = idx % NBQ;
  const int h = (idx / NBQ) % HQ;
  const int b = idx / (NBQ * HQ);
  const int tid = threadIdx.x, lane = tid & 31, w = tid >> 5;
  const int qbase = w * 16;
  const int hs8 = (lane >> 4) << 3;         // 0 or 8 (row/k half select)

  const size_t headQK = (size_t)(b * HQ + h) * SQ * DQ;
  const size_t headV  = (size_t)(b * HQ + h) * DQ * SQ;

  // staging coordinates for the async V^T copy
  const int vd = tid >> 2;
  const int vc0 = (tid & 3) * 32;

  // Q fragments for this wave's 16 rows (two k-chunks of 32 over D=64)
  v16h qf[2];
  {
    const _Float16* qrow =
        Qh + headQK + (size_t)(n * BLKQ + qbase + (lane & 15)) * DQ;
#pragma unroll
    for (int kc = 0; kc < 2; ++kc) {
      const _Float16* p = qrow + kc * 32 + hs8;
      v8h* hh = (v8h*)&qf[kc];
      hh[0] = *(const v8h*)(p);
      hh[1] = *(const v8h*)(p + 16);
    }
  }

  float m_r[8], l_r[8];
  v8f oacc[4];
#pragma unroll
  for (int r = 0; r < 8; ++r) { m_r[r] = -1e30f; l_r[r] = 0.0f; }
#pragma unroll
  for (int nt = 0; nt < 4; ++nt) oacc[nt] = vzero8();

  const int j0 = (n > 0) ? n - 1 : n;
  const int j1 = (n < NBQ - 1) ? n + 1 : n;

  // issue loads for the first key block into buffer 0
  if (w == 0) tdm_load_k(Kh + headQK + (size_t)j0 * BLKQ * DQ, Ks);
  {
    const _Float16* src = Vth + headV + (size_t)vd * SQ + j0 * BLKQ + vc0;
    _Float16* dst = Vs + vd * VLD + vc0;
#pragma unroll
    for (int i = 0; i < 32; i += 8) async_cp16(dst + i, src + i);
  }

  int cur = 0;
  for (int jb = j0; jb <= j1; ++jb) {
    WAIT_ASYNC0();
    __builtin_amdgcn_s_wait_tensorcnt(0);
    __syncthreads();                           // buffer `cur` ready for all waves

    // issue next block's DMA into the other buffer (overlaps with compute)
    if (jb < j1) {
      if (w == 0)
        tdm_load_k(Kh + headQK + (size_t)(jb + 1) * BLKQ * DQ, Ks + (cur ^ 1) * KSZ);
      const _Float16* src = Vth + headV + (size_t)vd * SQ + (jb + 1) * BLKQ + vc0;
      _Float16* dst = Vs + (cur ^ 1) * VSZ + vd * VLD + vc0;
#pragma unroll
      for (int i = 0; i < 32; i += 8) async_cp16(dst + i, src + i);
    }

    const _Float16* Kb = Ks + cur * KSZ;
    const _Float16* Vb = Vs + cur * VSZ;

    // ---- scores S = Q K^T (16 x 128 per wave) ----
    v8f sacc[8];
#pragma unroll
    for (int kt = 0; kt < 8; ++kt) sacc[kt] = vzero8();
#pragma unroll
    for (int kt = 0; kt < 8; ++kt) {
#pragma unroll
      for (int kc = 0; kc < 2; ++kc) {
        v16h bf = ld_bfrag(Kb + (kt * 16) * KLD + kc * 32, KLD, lane);
        sacc[kt] = __builtin_amdgcn_wmma_f32_16x16x32_f16(
            false, qf[kc], false, bf, (short)0, sacc[kt], false, false);
      }
    }

    // ---- scale + band mask + block row-max ----
    const int rel = jb - n;                   // -1, 0, +1
    float bm[8];
#pragma unroll
    for (int r = 0; r < 8; ++r) bm[r] = -1e30f;
#pragma unroll
    for (int kt = 0; kt < 8; ++kt) {
      const int kp = kt * 16 + (lane & 15);
#pragma unroll
      for (int r = 0; r < 8; ++r) {
        const int qp = qbase + hs8 + r;
        float v = sacc[kt][r] * 0.125f;       // 1/sqrt(64)
        const bool keep = (rel == 0) || ((rel < 0) ? (kp >= qp) : (kp < qp));
        v = keep ? v : -1e30f;
        sacc[kt][r] = v;
        bm[r] = fmaxf(bm[r], v);
      }
    }
#pragma unroll
    for (int r = 0; r < 8; ++r)
#pragma unroll
      for (int xm = 1; xm < 16; xm <<= 1)
        bm[r] = fmaxf(bm[r], __shfl_xor(bm[r], xm, 32));

    // ---- online softmax update ----
    float scn[8];
#pragma unroll
    for (int r = 0; r < 8; ++r) {
      const float mn = fmaxf(m_r[r], bm[r]);
      scn[r] = __expf(m_r[r] - mn);
      m_r[r] = mn;
    }
    float bs[8];
#pragma unroll
    for (int r = 0; r < 8; ++r) bs[r] = 0.0f;
#pragma unroll
    for (int kt = 0; kt < 8; ++kt)
#pragma unroll
      for (int r = 0; r < 8; ++r) {
        const float p = __expf(sacc[kt][r] - m_r[r]);
        sacc[kt][r] = p;
        bs[r] += p;
      }
#pragma unroll
    for (int r = 0; r < 8; ++r)
#pragma unroll
      for (int xm = 1; xm < 16; xm <<= 1)
        bs[r] += __shfl_xor(bs[r], xm, 32);
#pragma unroll
    for (int r = 0; r < 8; ++r) l_r[r] = l_r[r] * scn[r] + bs[r];
#pragma unroll
    for (int nt = 0; nt < 4; ++nt)
#pragma unroll
      for (int r = 0; r < 8; ++r) oacc[nt][r] *= scn[r];

    // ---- relayout P (C-layout -> A-layout) via per-wave LDS tile ----
    _Float16* Pw = Ps + w * 16 * PLD;
#pragma unroll
    for (int kt = 0; kt < 8; ++kt) {
      const int kp = kt * 16 + (lane & 15);
#pragma unroll
      for (int r = 0; r < 8; ++r)
        Pw[(hs8 + r) * PLD + kp] = (_Float16)sacc[kt][r];
    }
    // (LDS ops are in-order within a wave; Pw is private to this wave.)

    // ---- O += P x V ----
#pragma unroll
    for (int kc = 0; kc < 4; ++kc) {
      v16h af = ld_afrag(Pw + kc * 32, PLD, lane);
#pragma unroll
      for (int nt = 0; nt < 4; ++nt) {
        v16h bf = ld_bfrag(Vb + (nt * 16) * VLD + kc * 32, VLD, lane);
        oacc[nt] = __builtin_amdgcn_wmma_f32_16x16x32_f16(
            false, af, false, bf, (short)0, oacc[nt], false, false);
      }
    }
    cur ^= 1;
  }

  // ---- normalize and store O as f16 [B,S,H,D] ----
#pragma unroll
  for (int r = 0; r < 8; ++r) l_r[r] = 1.0f / l_r[r];
#pragma unroll
  for (int nt = 0; nt < 4; ++nt) {
    const int d = nt * 16 + (lane & 15);
#pragma unroll
    for (int r = 0; r < 8; ++r) {
      const int q = n * BLKQ + qbase + hs8 + r;
      const size_t addr = ((size_t)(b * SQ + q) * HQ + h) * DQ + d;
      Oh[addr] = (_Float16)(oacc[nt][r] * l_r[r]);
    }
  }
}

// ---------------------------------------------------------------------------
// Stage C: output projection.  out = O[16384,512](f16) x Wo[512,512] + bo (f32)
// A tile: async global->LDS (pure f16 copy), double-buffered; B tile staged
// through VGPRs with the same one-barrier pipeline.
// ---------------------------------------------------------------------------
__global__ __launch_bounds__(256)
void out_proj_kernel(const _Float16* __restrict__ Oh, const float* __restrict__ Wo,
                     const float* __restrict__ bo, float* __restrict__ out) {
  constexpr int LDK = 40;
  constexpr int TSZ = 128 * LDK;
  __shared__ _Float16 As[2 * TSZ];
  __shared__ _Float16 Bs[2 * TSZ];

  const int m0 = blockIdx.x * 128;
  const int n0 = blockIdx.y * 128;
  const int tid = threadIdx.x;
  const int lane = tid & 31;
  const int w  = tid >> 5;
  const int wm = w & 3;
  const int wn = w >> 2;

  const int arow = tid >> 1, ac0 = (tid & 1) * 16;
  const int bk_ = tid >> 3,  bn0 = (tid & 7) * 16;
  const _Float16* aptr = Oh + (size_t)(m0 + arow) * FQ + ac0;
  const float*    bptr = Wo + (size_t)bk_ * FQ + n0 + bn0;

  v8f acc[2][4];
#pragma unroll
  for (int i = 0; i < 2; ++i)
#pragma unroll
    for (int j = 0; j < 4; ++j) acc[i][j] = vzero8();

  float breg[16];

  // prologue: async A chunk 0; fetch + stage B chunk 0 (buffer 0)
  {
    _Float16* Ad = As + arow * LDK + ac0;
    async_cp16(Ad, aptr);
    async_cp16(Ad + 8, aptr + 8);
    load16f(breg, bptr);
#pragma unroll
    for (int i = 0; i < 16; ++i) Bs[(bn0 + i) * LDK + bk_] = (_Float16)breg[i];
  }

  int cur = 0;
  for (int kc = 0; kc < FQ; kc += 32) {
    WAIT_ASYNC0();
    __syncthreads();                            // buffer `cur` ready
    const bool have_next = (kc + 32 < FQ);
    if (have_next) {
      // async A for next chunk into other buffer; B loads into VGPRs
      _Float16* Ad = As + (cur ^ 1) * TSZ + arow * LDK + ac0;
      const _Float16* asrc = aptr + kc + 32;
      async_cp16(Ad, asrc);
      async_cp16(Ad + 8, asrc + 8);
      load16f(breg, bptr + (size_t)(kc + 32) * FQ);
    }

    const _Float16* Ab = As + cur * TSZ;
    const _Float16* Bb = Bs + cur * TSZ;
    v16h bfr[4];
#pragma unroll
    for (int nt = 0; nt < 4; ++nt)
      bfr[nt] = ld_bfrag(Bb + (wn * 64 + nt * 16) * LDK, LDK, lane);
#pragma unroll
    for (int mt = 0; mt < 2; ++mt) {
      v16h afr = ld_afrag(Ab + (wm * 32 + mt * 16) * LDK, LDK, lane);
#pragma unroll
      for (int nt = 0; nt < 4; ++nt)
        acc[mt][nt] = __builtin_amdgcn_wmma_f32_16x16x32_f16(
            false, afr, false, bfr[nt], (short)0, acc[mt][nt], false, false);
    }

    if (have_next) {
      _Float16* Bd = Bs + (cur ^ 1) * TSZ;
#pragma unroll
      for (int i = 0; i < 16; ++i) Bd[(bn0 + i) * LDK + bk_] = (_Float16)breg[i];
    }
    cur ^= 1;
  }

#pragma unroll
  for (int nt = 0; nt < 4; ++nt) {
    const int nn = n0 + wn * 64 + nt * 16 + (lane & 15);
    const float bv_ = bo[nn];
#pragma unroll
    for (int mt = 0; mt < 2; ++mt) {
#pragma unroll
      for (int r = 0; r < 8; ++r) {
        const int m = m0 + wm * 32 + mt * 16 + ((lane >> 4) << 3) + r;
        out[(size_t)m * FQ + nn] = acc[mt][nt][r] + bv_;
      }
    }
  }
}

// ---------------------------------------------------------------------------
extern "C" void kernel_launch(void* const* d_in, const int* in_sizes, int n_in,
                              void* d_out, int out_size, void* d_ws, size_t ws_size,
                              hipStream_t stream) {
  (void)in_sizes; (void)n_in; (void)out_size; (void)ws_size;
  const float* inputs_q  = (const float*)d_in[0];
  const float* inputs_kv = (const float*)d_in[1];
  const float* Wq = (const float*)d_in[2];
  const float* bq = (const float*)d_in[3];
  const float* Wk = (const float*)d_in[4];
  const float* bk = (const float*)d_in[5];
  const float* Wv = (const float*)d_in[6];
  const float* bv = (const float*)d_in[7];
  const float* Wo = (const float*)d_in[8];
  const float* bo = (const float*)d_in[9];
  float* out = (float*)d_out;

  // Workspace: Q, K (f16 [B,H,S,D]), V^T (f16 [B,H,D,S]), O (f16 [B,S,H,D])
  // = 4 * 2*8*8192*64 halfs = 64 MB total.
  const size_t NE = (size_t)BQ * HQ * SQ * DQ;
  _Float16* Qh  = (_Float16*)d_ws;
  _Float16* Kh  = Qh + NE;
  _Float16* Vth = Kh + NE;
  _Float16* Oh  = Vth + NE;

  qkv_proj_kernel<<<dim3(MROWS / 128, FQ / 128, 3), 256, 0, stream>>>(
      inputs_q, inputs_kv, Wq, bq, Wk, bk, Wv, bv, Qh, Kh, Vth);
  local_attn_kernel<<<dim3(BQ * HQ * NBQ), 256, 0, stream>>>(Qh, Kh, Vth, Oh);
  out_proj_kernel<<<dim3(MROWS / 128, FQ / 128), 256, 0, stream>>>(Oh, Wo, bo, out);
}
